// GCN_7773890806107
// MI455X (gfx1250) — compile-verified, matrix operand order checked by reading, not run on hardware
//
#include <hip/hip_runtime.h>
#include <hip/hip_bf16.h>

// ---------------------------------------------------------------------------
// GCN: h1 = ReLU(Agg(x @ W1) + b1); h2 = ReLU(Agg(h1 @ W2) + b2); out = h2@Wc + bc
// Agg(h)[i] = sum_{e: dst=i} h[src]*dinv[src]*dinv[i]  +  h[i]*dinv[i]^2
// deg[i] = indeg(i) + 1 (self loop), dinv = rsqrt(deg).
// GEMMs: bf16 WMMA (v_wmma_f32_16x16x32_bf16), weights staged to LDS via TDM.
// ---------------------------------------------------------------------------

#define FDIM 128

typedef __bf16  bf16_t;
typedef bf16_t  v8bf   __attribute__((ext_vector_type(8)));
typedef bf16_t  v16bf  __attribute__((ext_vector_type(16)));
typedef float   v8f    __attribute__((ext_vector_type(8)));
typedef unsigned int v4u  __attribute__((ext_vector_type(4)));
typedef int          v8i  __attribute__((ext_vector_type(8)));
typedef int          v4i  __attribute__((ext_vector_type(4)));

#if defined(__HIP_DEVICE_COMPILE__) && \
    __has_builtin(__builtin_amdgcn_tensor_load_to_lds) && \
    __has_builtin(__builtin_amdgcn_s_wait_tensorcnt)
#define USE_TDM 1
#else
#define USE_TDM 0
#endif

__device__ __forceinline__ unsigned short f32_to_bf16_bits(float f) {
    unsigned int u = __float_as_uint(f);
    unsigned int r = u + 0x7FFFu + ((u >> 16) & 1u);   // round-to-nearest-even
    return (unsigned short)(r >> 16);
}

// ---------------- degree / normalization ----------------
__global__ __launch_bounds__(256) void k_deg_init(float* deg, int n) {
    int i = blockIdx.x * blockDim.x + threadIdx.x;
    if (i < n) deg[i] = 1.0f;                          // self loop
}

__global__ __launch_bounds__(256) void k_deg_acc(const int* __restrict__ dst,
                                                 float* deg, int e) {
    int i = blockIdx.x * blockDim.x + threadIdx.x;
    if (i < e) atomicAdd(&deg[dst[i]], 1.0f);
}

__global__ __launch_bounds__(256) void k_deg_fin(float* deg, int n) {
    int i = blockIdx.x * blockDim.x + threadIdx.x;
    if (i < n) deg[i] = rsqrtf(deg[i]);
}

// ---------------- f32 -> bf16 conversions ----------------
__global__ __launch_bounds__(256) void k_cvt_bf16(const float* __restrict__ in,
                                                  unsigned short* __restrict__ out,
                                                  int count) {
    int i = blockIdx.x * blockDim.x + threadIdx.x;
    if (i < count) out[i] = f32_to_bf16_bits(in[i]);
}

// W[k][n] (row-major KxN) -> Wt[n][k] bf16 (so B-fragment loads are contiguous)
__global__ __launch_bounds__(256) void k_cvt_w_transpose(const float* __restrict__ W,
                                                         unsigned short* __restrict__ Wt) {
    int i = blockIdx.x * blockDim.x + threadIdx.x;   // 0 .. 128*128-1
    if (i >= FDIM * FDIM) return;
    int n = i & (FDIM - 1);
    int k = i >> 7;
    Wt[n * FDIM + k] = f32_to_bf16_bits(W[k * FDIM + n]);
}

// ---------------- bf16 WMMA GEMM: C[M,128] = A[M,128] @ W[128,128] ----------------
// A: bf16 bits, row-major [M][128].  Wt: bf16 bits, transposed [n][k].
// Wt is staged into LDS once per block (TDM if available, else cooperative copy).
// One wave = 16 rows x 128 cols (8 accum tiles). Block = 256 threads = 8 waves.
__global__ __launch_bounds__(256) void k_gemm_wmma(const unsigned short* __restrict__ A,
                                                   const unsigned short* __restrict__ Wt,
                                                   float* __restrict__ C, int M) {
    __shared__ __align__(16) unsigned short Ws[FDIM * FDIM];   // 32 KB

    const int tid  = threadIdx.x;
    const int lane = tid & 31;
    const int wave = tid >> 5;

#if USE_TDM
    if (wave == 0) {
        // --- Tensor DMA descriptor (CDNA5 ISA) ---
        // 2D tensor of 2-byte elements, 128x128, tile == whole tensor.
        unsigned int lds_base = (unsigned int)(unsigned long long)
            (__attribute__((address_space(3))) unsigned short*)&Ws[0];
        unsigned long long ga = (unsigned long long)Wt;
        v4u g0;
        g0[0] = 1u;                                           // count=1 (valid), load
        g0[1] = lds_base;                                     // lds_addr (bytes)
        g0[2] = (unsigned int)(ga & 0xFFFFFFFFu);             // global_addr[31:0]
        g0[3] = (unsigned int)((ga >> 32) & 0x1FFFFFFu)       // global_addr[56:32]
              | (2u << 30);                                   // type=2 ("image")
        v8i g1;
        g1[0] = (int)(1u << 16);       // workgroup_mask=0, data_size=1 (2 bytes)
        g1[1] = (int)(128u << 16);     // tensor_dim0[15:0]=128 in bits[31:16]
        g1[2] = (int)(128u << 16);     // tensor_dim0 hi=0; tensor_dim1 lo=128
        g1[3] = (int)(128u << 16);     // tensor_dim1 hi=0; tile_dim0=128
        g1[4] = 128;                   // tile_dim1=128; tile_dim2=0
        g1[5] = 128;                   // tensor_dim0_stride low32 = 128 elems
        g1[6] = 0;                     // stride0 hi / stride1 lo
        g1[7] = 0;
        v4i z4 = {0, 0, 0, 0};
#if __has_include(<hip/amd_detail/amd_gfx1250_TDM.h>)
        v8i z8 = {0, 0, 0, 0, 0, 0, 0, 0};
        __builtin_amdgcn_tensor_load_to_lds(g0, g1, z4, z4, z8, 0);  // 6-arg toolchain
#else
        __builtin_amdgcn_tensor_load_to_lds(g0, g1, z4, z4, 0);      // 5-arg toolchain
#endif
        __builtin_amdgcn_s_wait_tensorcnt(0);
    }
    __syncthreads();
#else
    {   // cooperative stage-in: 2048 uint4 / 256 threads = 8 each
        const uint4* s4 = (const uint4*)Wt;
        uint4*       d4 = (uint4*)Ws;
        #pragma unroll
        for (int i = 0; i < 8; ++i) d4[tid + i * 256] = s4[tid + i * 256];
        __syncthreads();
    }
#endif

    const int r0 = blockIdx.x * 128 + wave * 16;
    if (r0 >= M) return;                   // wave-uniform: EXEC stays all-1s for WMMA

    const int mrow = lane & 15;            // row within tile (A) / col within tile (B)
    const int koff = (lane >> 4) << 3;     // 0 or 8: K-group per documented 16-bit layout

    v8f acc[8];
    #pragma unroll
    for (int t = 0; t < 8; ++t) acc[t] = (v8f){0.f,0.f,0.f,0.f,0.f,0.f,0.f,0.f};

    const unsigned short* arow = A + (size_t)(r0 + mrow) * FDIM;

    #pragma unroll
    for (int kc = 0; kc < FDIM; kc += 32) {
        // A fragment: K = kc+koff..+7 -> elems 0..7 (VGPR0-3); +16..+23 -> elems 8..15
        v8bf alo = *(const v8bf*)(arow + kc + koff);
        v8bf ahi = *(const v8bf*)(arow + kc + koff + 16);
        v16bf a;
        #pragma unroll
        for (int i = 0; i < 8; ++i) { a[i] = alo[i]; a[i + 8] = ahi[i]; }

        #pragma unroll
        for (int t = 0; t < 8; ++t) {
            const unsigned short* brow = Ws + (t * 16 + mrow) * FDIM;   // LDS
            v8bf blo = *(const v8bf*)(brow + kc + koff);
            v8bf bhi = *(const v8bf*)(brow + kc + koff + 16);
            v16bf b;
            #pragma unroll
            for (int i = 0; i < 8; ++i) { b[i] = blo[i]; b[i + 8] = bhi[i]; }

            acc[t] = __builtin_amdgcn_wmma_f32_16x16x32_bf16(
                false, a, false, b, (short)0, acc[t], false, false);
        }
    }

    // C/D layout: lane<16 -> rows v+0, lane>=16 -> rows v+8; col = lane&15
    const int rowadd = (lane >> 4) * 8;
    const int col    = lane & 15;
    #pragma unroll
    for (int t = 0; t < 8; ++t) {
        float* cb = C + (size_t)(r0 + rowadd) * FDIM + t * 16 + col;
        #pragma unroll
        for (int v = 0; v < 8; ++v) cb[(size_t)v * FDIM] = acc[t][v];
    }
}

// ---------------- aggregation ----------------
// Self-loop term also initializes the output buffer (no memset needed).
__global__ __launch_bounds__(256) void k_agg_self(const float* __restrict__ h,
                                                  const float* __restrict__ dinv,
                                                  float* __restrict__ out, int n) {
    int idx = blockIdx.x * blockDim.x + threadIdx.x;   // n * 32 threads (float4 chunks)
    if (idx >= n * 32) return;
    int node = idx >> 5, c = idx & 31;
    float w = dinv[node]; w = w * w;
    float4 v = ((const float4*)(h + (size_t)node * FDIM))[c];
    ((float4*)(out + (size_t)node * FDIM))[c] =
        make_float4(v.x * w, v.y * w, v.z * w, v.w * w);
}

__global__ __launch_bounds__(256) void k_agg_edges(const int* __restrict__ src,
                                                   const int* __restrict__ dst,
                                                   const float* __restrict__ dinv,
                                                   const float* __restrict__ h,
                                                   float* __restrict__ out, int e) {
    int idx = blockIdx.x * blockDim.x + threadIdx.x;   // e * 32 threads
    if (idx >= e * 32) return;
    int ed = idx >> 5, c = idx & 31;
    int s = src[ed], d = dst[ed];
    float w = dinv[s] * dinv[d];
    float4 v = ((const float4*)(h + (size_t)s * FDIM))[c];
    float* o = out + (size_t)d * FDIM + c * 4;
    atomicAdd(o + 0, v.x * w);
    atomicAdd(o + 1, v.y * w);
    atomicAdd(o + 2, v.z * w);
    atomicAdd(o + 3, v.w * w);
}

// bias + ReLU; emit bf16 (feeds next WMMA GEMM)
__global__ __launch_bounds__(256) void k_bias_relu_bf16(const float* __restrict__ agg,
                                                        const float* __restrict__ bias,
                                                        unsigned short* __restrict__ outb,
                                                        int n) {
    int idx = blockIdx.x * blockDim.x + threadIdx.x;
    if (idx >= n * FDIM) return;
    float v = agg[idx] + bias[idx & (FDIM - 1)];
    v = v > 0.0f ? v : 0.0f;
    outb[idx] = f32_to_bf16_bits(v);
}

// bias + ReLU; emit f32 (feeds classifier)
__global__ __launch_bounds__(256) void k_bias_relu_f32(const float* __restrict__ agg,
                                                       const float* __restrict__ bias,
                                                       float* __restrict__ outf, int n) {
    int idx = blockIdx.x * blockDim.x + threadIdx.x;
    if (idx >= n * FDIM) return;
    float v = agg[idx] + bias[idx & (FDIM - 1)];
    outf[idx] = v > 0.0f ? v : 0.0f;
}

// ---------------- classifier: out[n,10] = h[n,128] @ Wc[128,10] + bc ----------------
__global__ __launch_bounds__(256) void k_classifier(const float* __restrict__ h,
                                                    const float* __restrict__ Wc,
                                                    const float* __restrict__ bc,
                                                    float* __restrict__ out, int n) {
    int node = blockIdx.x * blockDim.x + threadIdx.x;
    if (node >= n) return;
    float acc[10];
    #pragma unroll
    for (int c = 0; c < 10; ++c) acc[c] = bc[c];
    const float* row = h + (size_t)node * FDIM;
    for (int k = 0; k < FDIM; ++k) {
        float xv = row[k];
        #pragma unroll
        for (int c = 0; c < 10; ++c) acc[c] += xv * Wc[k * 10 + c];
    }
    #pragma unroll
    for (int c = 0; c < 10; ++c) out[(size_t)node * 10 + c] = acc[c];
}

// ---------------------------------------------------------------------------
extern "C" void kernel_launch(void* const* d_in, const int* in_sizes, int n_in,
                              void* d_out, int out_size, void* d_ws, size_t ws_size,
                              hipStream_t stream) {
    const float* x   = (const float*)d_in[0];
    const int*   ei  = (const int*)  d_in[1];
    const float* W1  = (const float*)d_in[2];
    const float* b1  = (const float*)d_in[3];
    const float* W2  = (const float*)d_in[4];
    const float* b2  = (const float*)d_in[5];
    const float* Wc  = (const float*)d_in[6];
    const float* bc  = (const float*)d_in[7];
    float*       out = (float*)d_out;

    const int N = in_sizes[0] / FDIM;     // 50000
    const int E = in_sizes[1] / 2;        // 800000
    const int* src = ei;
    const int* dst = ei + E;

    // workspace carving (256B aligned)
    unsigned char* wsb = (unsigned char*)d_ws;
    size_t off = 0;
    auto take = [&](size_t bytes) -> void* {
        void* p = wsb + off;
        off += (bytes + 255) & ~(size_t)255;
        return p;
    };
    float*          dinv = (float*)         take((size_t)N * 4);
    float*          bufA = (float*)         take((size_t)N * FDIM * 4); // GEMM out / h2 f32
    float*          bufB = (float*)         take((size_t)N * FDIM * 4); // agg accumulator
    unsigned short* bufC = (unsigned short*)take((size_t)N * FDIM * 2); // bf16 GEMM input
    unsigned short* W1b  = (unsigned short*)take(FDIM * FDIM * 2);
    unsigned short* W2b  = (unsigned short*)take(FDIM * FDIM * 2);

    const int T = 256;
    auto blk = [&](long long work) { return (int)((work + T - 1) / T); };

    // normalization
    k_deg_init<<<blk(N), T, 0, stream>>>(dinv, N);
    k_deg_acc <<<blk(E), T, 0, stream>>>(dst, dinv, E);
    k_deg_fin <<<blk(N), T, 0, stream>>>(dinv, N);

    // weight prep (transposed bf16)
    k_cvt_w_transpose<<<blk(FDIM * FDIM), T, 0, stream>>>(W1, W1b);
    k_cvt_w_transpose<<<blk(FDIM * FDIM), T, 0, stream>>>(W2, W2b);

    // ---- layer 1 ----
    k_cvt_bf16<<<blk((long long)N * FDIM), T, 0, stream>>>(x, bufC, N * FDIM);
    k_gemm_wmma<<<(N + 127) / 128, 256, 0, stream>>>(bufC, W1b, bufA, N);
    k_agg_self <<<blk((long long)N * 32), T, 0, stream>>>(bufA, dinv, bufB, N);
    k_agg_edges<<<blk((long long)E * 32), T, 0, stream>>>(src, dst, dinv, bufA, bufB, E);
    k_bias_relu_bf16<<<blk((long long)N * FDIM), T, 0, stream>>>(bufB, b1, bufC, N);

    // ---- layer 2 ----
    k_gemm_wmma<<<(N + 127) / 128, 256, 0, stream>>>(bufC, W2b, bufA, N);
    k_agg_self <<<blk((long long)N * 32), T, 0, stream>>>(bufA, dinv, bufB, N);
    k_agg_edges<<<blk((long long)E * 32), T, 0, stream>>>(src, dst, dinv, bufA, bufB, E);
    k_bias_relu_f32<<<blk((long long)N * FDIM), T, 0, stream>>>(bufB, b2, bufA, N);

    // ---- classifier ----
    k_classifier<<<blk(N), T, 0, stream>>>(bufA, Wc, bc, out, N);
}
